// Heads_50079318672017
// MI455X (gfx1250) — compile-verified, matrix operand order checked by reading, not run on hardware
//
#include <hip/hip_runtime.h>

typedef __attribute__((ext_vector_type(16))) _Float16 v16h;
typedef __attribute__((ext_vector_type(8)))  _Float16 h8;
typedef __attribute__((ext_vector_type(4)))  _Float16 h4;
typedef __attribute__((ext_vector_type(8)))  float    v8f;
typedef __attribute__((ext_vector_type(4)))  unsigned u32x4;
typedef __attribute__((ext_vector_type(8)))  unsigned u32x8;

constexpr int kH  = 16;
constexpr int kDK = 64;
constexpr int kDM = 1024;
constexpr int kT  = 2048;
constexpr int kB  = 4;
constexpr int kTiles = kT / 64;   // 32 key tiles per (b,h)

union F16x16 { v16h v; h8 h[2]; };

// Build a 16-element f16 WMMA fragment from two contiguous 8-half (16B) chunks.
// CDNA5 16-bit A/B layout: elements 0-7 -> K = half*8 + 0..7,
// elements 8-15 -> K = 16 + half*8 + 0..7.
static __device__ __forceinline__ v16h ld_frag(const _Float16* p0) {
  F16x16 f;
  f.h[0] = *(const h8*)(p0);
  f.h[1] = *(const h8*)(p0 + 16);
  return f.v;
}

// ---------------------------------------------------------------------------
// TDM: issue a 2D f16 tile load Global -> LDS via the Tensor Data Mover.
// Tile: `rows` x `rowlen` halves, row stride in memory = row_stride_elems.
// LDS destination rows are padded 128B data + 16B pad (72-half row stride):
//   pad_interval code 4 = 32 DWORDs (128B), pad_amount code 3 = 4 DWORDs (16B).
// Descriptor layout per CDNA5 ISA ch.8 (D# group0: 4 SGPRs, group1: 8 SGPRs).
// ---------------------------------------------------------------------------
static __device__ __forceinline__ void tdm_load_2d(
    const _Float16* gptr, unsigned lds_off, unsigned rows, unsigned rowlen,
    unsigned row_stride_elems)
{
  unsigned long long ga = (unsigned long long)(size_t)gptr;
  u32x4 g0;
  g0[0] = 1u;                                                // count=1, user D#
  g0[1] = lds_off;                                           // lds_addr (bytes)
  g0[2] = (unsigned)ga;                                      // global_addr[31:0]
  g0[3] = (unsigned)((ga >> 32) & 0x1FFFFFFu) | (2u << 30);  // addr[56:32]|type=2
  u32x8 g1;
  g1[0] = (1u << 16)      // data_size = 2 bytes
        | (1u << 20)      // pad_enable
        | (4u << 22)      // pad_interval: 32 DWORDs
        | (3u << 25);     // pad_amount: 4 DWORDs
  g1[1] = (rowlen & 0xFFFFu) << 16;   // tensor_dim0[15:0]  (bits 79:48, lo half)
  g1[2] = (rows   & 0xFFFFu) << 16;   // tensor_dim1[15:0]  (bits 111:80, lo half)
  g1[3] = (rowlen & 0xFFFFu) << 16;   // tile_dim0          (bits 127:112)
  g1[4] = (rows   & 0xFFFFu);         // tile_dim1; tile_dim2 = 0
  g1[5] = row_stride_elems;           // tensor_dim0_stride[31:0]
  g1[6] = 0u;                         // stride0 hi | tensor_dim1_stride lo
  g1[7] = 0u;                         // tensor_dim1_stride hi
  asm volatile("tensor_load_to_lds %0, %1" :: "s"(g0), "s"(g1) : "memory");
}

// ---------------------------------------------------------------------------
// Kernel 1: Y = X @ W^T + b  (per z: q/k/v), f16 WMMA with f32 accumulate.
// Output f16 to workspace: q,k -> [B,H,T,DK]; v -> [B,H,DK,T] (transposed).
// q pre-scaled by 1/sqrt(DK) = 0.125.
// ---------------------------------------------------------------------------
__global__ __launch_bounds__(256) void proj_kernel(
    const float* __restrict__ Q, const float* __restrict__ K, const float* __restrict__ V,
    const float* __restrict__ Wq, const float* __restrict__ bq,
    const float* __restrict__ Wk, const float* __restrict__ bk,
    const float* __restrict__ Wv, const float* __restrict__ bv,
    _Float16* __restrict__ q16, _Float16* __restrict__ k16, _Float16* __restrict__ vT)
{
  const int z = blockIdx.z;
  const float* X    = (z == 0) ? Q  : (z == 1) ? K  : V;
  const float* W    = (z == 0) ? Wq : (z == 1) ? Wk : Wv;
  const float* bias = (z == 0) ? bq : (z == 1) ? bk : bv;
  _Float16* out     = (z == 0) ? q16 : (z == 1) ? k16 : vT;

  __shared__ _Float16 Xs[128][40];   // 32 K-cols + 8 pad halves (16B aligned rows)
  __shared__ _Float16 Ws[64][40];

  const int tid  = threadIdx.x;
  const int lane = tid & 31;
  const int wid  = tid >> 5;
  const int l16  = lane & 15;
  const int hf   = lane >> 4;
  const int wave_m = wid & 3;    // 4 waves along M (32 rows each)
  const int wave_n = wid >> 2;   // 2 waves along N (32 cols each)
  const int m0 = blockIdx.y * 128;
  const int n0 = blockIdx.x * 64;

  v8f zero = {};
  v8f c[2][2];
  #pragma unroll
  for (int i = 0; i < 2; i++)
    #pragma unroll
    for (int j = 0; j < 2; j++) c[i][j] = zero;

  for (int kb = 0; kb < kDM; kb += 32) {
    // Prefetch next K-slab with WGP scope (locality 3 -> near-cache prefetch)
    // so it lands in WGP$ while this slab is consumed by the WMMAs.
    if (kb + 32 < kDM) {
      __builtin_prefetch(&X[(size_t)(m0 + (tid >> 1)) * kDM + kb + 32 + (tid & 1) * 16], 0, 3);
      if (tid < 128)
        __builtin_prefetch(&W[(size_t)(n0 + (tid >> 1)) * kDM + kb + 32 + (tid & 1) * 16], 0, 3);
    }
    // Stage X tile (128x32 f32 -> f16): 1024 float4 chunks
    #pragma unroll
    for (int it = 0; it < 4; it++) {
      int ci  = tid + it * 256;
      int row = ci >> 3;
      int c4  = (ci & 7) * 4;
      float4 xv = *(const float4*)&X[(size_t)(m0 + row) * kDM + kb + c4];
      h4 hv = { (_Float16)xv.x, (_Float16)xv.y, (_Float16)xv.z, (_Float16)xv.w };
      *(h4*)&Xs[row][c4] = hv;
    }
    // Stage W tile (64x32 f32 -> f16): rows of W along d (B[k][n] = W[n][k])
    #pragma unroll
    for (int it = 0; it < 2; it++) {
      int ci  = tid + it * 256;
      int row = ci >> 3;
      int c4  = (ci & 7) * 4;
      float4 wv = *(const float4*)&W[(size_t)(n0 + row) * kDM + kb + c4];
      h4 hv = { (_Float16)wv.x, (_Float16)wv.y, (_Float16)wv.z, (_Float16)wv.w };
      *(h4*)&Ws[row][c4] = hv;
    }
    __syncthreads();

    v16h a[2], bfr[2];
    #pragma unroll
    for (int mi = 0; mi < 2; mi++)
      a[mi] = ld_frag(&Xs[wave_m * 32 + mi * 16 + l16][hf * 8]);
    #pragma unroll
    for (int ni = 0; ni < 2; ni++)
      bfr[ni] = ld_frag(&Ws[wave_n * 32 + ni * 16 + l16][hf * 8]);

    #pragma unroll
    for (int mi = 0; mi < 2; mi++)
      #pragma unroll
      for (int ni = 0; ni < 2; ni++)
        c[mi][ni] = __builtin_amdgcn_wmma_f32_16x16x32_f16(
            false, a[mi], false, bfr[ni], (short)0, c[mi][ni], false, false);
    __syncthreads();
  }

  // Bias + (scale for q) + store f16 to workspace layouts.
  #pragma unroll
  for (int ni = 0; ni < 2; ni++) {
    int gc = n0 + wave_n * 32 + ni * 16 + l16;          // 0..1023
    float bval = bias[gc];
    int hh = gc >> 6, dk = gc & 63;
    #pragma unroll
    for (int mi = 0; mi < 2; mi++) {
      #pragma unroll
      for (int r = 0; r < 8; r++) {
        int gm = m0 + wave_m * 32 + mi * 16 + r + hf * 8;  // 0..8191
        int bb = gm >> 11, t = gm & 2047;
        int bh = bb * kH + hh;
        float val = c[mi][ni][r] + bval;
        if (z == 0) val *= 0.125f;                      // 1/sqrt(DK)
        size_t idx = (z == 2) ? ((size_t)bh * kDK + dk) * kT + t
                              : ((size_t)bh * kT + t) * kDK + dk;
        out[idx] = (_Float16)val;
      }
    }
  }
}

// ---------------------------------------------------------------------------
// Kernel 2: flash attention per (b,h). Block = 128 queries (8 waves x 16 rows),
// key tiles of 64. K/V tiles staged by the Tensor Data Mover into
// double-buffered LDS (software pipelined, TENSORcnt-synchronized).
// QK^T and PV via v_wmma_f32_16x16x32_f16, online softmax.
// ---------------------------------------------------------------------------
__global__ __launch_bounds__(256) void attn_kernel(
    const _Float16* __restrict__ q16, const _Float16* __restrict__ k16,
    const _Float16* __restrict__ vT, float* __restrict__ out)
{
  __shared__ _Float16 Ks[2][64][72];     // [buf][key][dk], TDM-padded rows
  __shared__ _Float16 VsT[2][64][72];    // [buf][dk][key], TDM-padded rows
  __shared__ _Float16 Ps[8][16][72];     // per-wave P staging [row][key]

  const int tid  = threadIdx.x;
  const int lane = tid & 31;
  const int wid  = tid >> 5;
  const int l16  = lane & 15;
  const int hf   = lane >> 4;
  const int bh   = blockIdx.y;
  const int b    = bh >> 4;
  const int hh   = bh & 15;
  const int q0   = blockIdx.x * 128;
  const int qrow = q0 + wid * 16 + l16;

  const _Float16* kbase = k16 + (size_t)bh * kT * kDK;   // [t][dk] rows
  const _Float16* vbase = vT  + (size_t)bh * kDK * kT;   // [dk][t] rows

  // One wave issues both TDM tile loads for `tile` into buffer tile&1.
  auto issue_tile = [&](int tile) {
    if (wid == 0) {
      int buf = tile & 1;
      tdm_load_2d(kbase + (size_t)tile * 64 * kDK,
                  (unsigned)(size_t)&Ks[buf][0][0], 64, 64, kDK);
      tdm_load_2d(vbase + tile * 64,
                  (unsigned)(size_t)&VsT[buf][0][0], 64, 64, kT);
    }
  };

  // Q A-fragments (dk = 64 -> 2 WMMA K-steps), kept in registers.
  v16h aq[2];
  #pragma unroll
  for (int ks = 0; ks < 2; ks++)
    aq[ks] = ld_frag(&q16[((size_t)bh * kT + qrow) * kDK + ks * 32 + hf * 8]);

  v8f zero = {};
  float mrow[8], lrow[8];
  v8f o[4];
  #pragma unroll
  for (int r = 0; r < 8; r++) { mrow[r] = -3.0e38f; lrow[r] = 0.f; }
  #pragma unroll
  for (int nt = 0; nt < 4; nt++) o[nt] = zero;

  // Software pipeline: tiles 0 and 1 in flight before the loop.
  issue_tile(0);
  issue_tile(1);

  for (int tile = 0; tile < kTiles; ++tile) {
    const int buf = tile & 1;
    // Each tile = 2 TDM ops, in-order completion: <=2 outstanding means the
    // oldest tile has fully landed; last tile must drain completely.
    if (tile == kTiles - 1) __builtin_amdgcn_s_wait_tensorcnt(0);
    else                    __builtin_amdgcn_s_wait_tensorcnt(2);
    __syncthreads();   // all waves see wave0's completed TDM data

    // S = Q K^T : 16 rows x 64 keys per wave (4 N-tiles, 2 K-steps).
    v8f s[4];
    #pragma unroll
    for (int nt = 0; nt < 4; nt++) {
      s[nt] = zero;
      #pragma unroll
      for (int ks = 0; ks < 2; ks++) {
        v16h bk = ld_frag(&Ks[buf][nt * 16 + l16][ks * 32 + hf * 8]);
        s[nt] = __builtin_amdgcn_wmma_f32_16x16x32_f16(
            false, aq[ks], false, bk, (short)0, s[nt], false, false);
      }
    }

    // Online softmax. C-layout: VGPR r holds row (r + hf*8), N = nt*16 + l16.
    #pragma unroll
    for (int r = 0; r < 8; r++) {
      float v = s[0][r];
      #pragma unroll
      for (int nt = 1; nt < 4; nt++) v = fmaxf(v, s[nt][r]);
      #pragma unroll
      for (int off = 1; off < 16; off <<= 1)
        v = fmaxf(v, __shfl_xor(v, off, 32));       // reduce within 16-lane half
      float mnew = fmaxf(mrow[r], v);
      float corr = __expf(mrow[r] - mnew);
      mrow[r] = mnew;
      lrow[r] *= corr;
      #pragma unroll
      for (int nt = 0; nt < 4; nt++) o[nt][r] *= corr;
      float psum = 0.f;
      #pragma unroll
      for (int nt = 0; nt < 4; nt++) {
        float p = __expf(s[nt][r] - mnew);
        psum += p;
        Ps[wid][r + hf * 8][nt * 16 + l16] = (_Float16)p;
      }
      #pragma unroll
      for (int off = 1; off < 16; off <<= 1)
        psum += __shfl_xor(psum, off, 32);
      lrow[r] += psum;
    }

    // Wave-private LDS RAW: drain DS stores before fragment re-load.
    asm volatile("s_wait_dscnt 0" ::: "memory");

    // O += P @ V : A = P (16x64 keys), B = V^T-staged (keys x dk).
    #pragma unroll
    for (int ks = 0; ks < 2; ks++) {
      v16h ap = ld_frag(&Ps[wid][l16][ks * 32 + hf * 8]);
      #pragma unroll
      for (int nt = 0; nt < 4; nt++) {
        v16h bv = ld_frag(&VsT[buf][nt * 16 + l16][ks * 32 + hf * 8]);
        o[nt] = __builtin_amdgcn_wmma_f32_16x16x32_f16(
            false, ap, false, bv, (short)0, o[nt], false, false);
      }
    }

    __syncthreads();   // all waves done reading buf before TDM overwrites it
    if (tile + 2 < kTiles) issue_tile(tile + 2);
  }

  // Finalize: O / l, write f32 to [B, T, D] with D-col = h*64 + dk.
  #pragma unroll
  for (int nt = 0; nt < 4; nt++) {
    int col = hh * kDK + nt * 16 + l16;
    #pragma unroll
    for (int r = 0; r < 8; r++) {
      int t = q0 + wid * 16 + r + hf * 8;
      out[((size_t)b * kT + t) * kDM + col] = o[nt][r] / lrow[r];
    }
  }
}

extern "C" void kernel_launch(void* const* d_in, const int* in_sizes, int n_in,
                              void* d_out, int out_size, void* d_ws, size_t ws_size,
                              hipStream_t stream) {
  const float* Q  = (const float*)d_in[0];
  const float* K  = (const float*)d_in[1];
  const float* V  = (const float*)d_in[2];
  const float* Wq = (const float*)d_in[3];
  const float* bq = (const float*)d_in[4];
  const float* Wk = (const float*)d_in[5];
  const float* bk = (const float*)d_in[6];
  const float* Wv = (const float*)d_in[7];
  const float* bv = (const float*)d_in[8];
  float* out = (float*)d_out;

  const size_t NHALF = (size_t)kB * kT * kDM;   // 8,388,608 f16 per tensor
  _Float16* q16 = (_Float16*)d_ws;
  _Float16* k16 = q16 + NHALF;
  _Float16* vTp = k16 + NHALF;

  dim3 g1(kDM / 64, (kB * kT) / 128, 3);
  proj_kernel<<<g1, 256, 0, stream>>>(Q, K, V, Wq, bq, Wk, bk, Wv, bv, q16, k16, vTp);

  dim3 g2(kT / 128, kB * kH);
  attn_kernel<<<g2, 256, 0, stream>>>(q16, k16, vTp, out);
}
